// BSquareModelCombined_91139206021417
// MI455X (gfx1250) — matrix operand order repaired and checked by reading, not compiled, not run on hardware
//
#include <hip/hip_runtime.h>
#include <stdint.h>

// ---------------------------------------------------------------------------
// SNN forward (LIF, 3 FC layers, T=100) for MI455X / gfx1250.
// bf16 weights (L2-resident: ~94MB < 192MB L2), f32-accumulate WMMA.
// Per block: 8 waves = 8 N-tiles share one 32xK A tile staged in LDS via
// async global->LDS copies (ASYNCcnt), double-buffered 2 stages deep.
// Weights are preloaded per stage into registers (one clause), then 8 WMMAs.
// ---------------------------------------------------------------------------

typedef __attribute__((ext_vector_type(16))) __bf16 v16bf;
typedef __attribute__((ext_vector_type(8)))  float  v8f;
typedef int v4i __attribute__((vector_size(16)));

union V16 { uint4 q[2]; v16bf v; };

#if defined(__has_builtin)
#if __has_builtin(__builtin_amdgcn_global_load_async_to_lds_b128)
#define USE_ASYNC 1
#endif
#endif
#ifndef USE_ASYNC
#define USE_ASYNC 0
#endif

// Stage geometry: Kc = 128 bf16 per stage, 32 rows, LDS row stride 136
// (272 B, 16B-aligned, spreads banks). Two buffers.
#define KC        128
#define SROW      136
#define BUFE      (32 * SROW)         // ushorts per buffer (4352)

__device__ __forceinline__ void async_cp_b128(const unsigned short* g,
                                              unsigned short* l) {
#if USE_ASYNC
  // Builtin signature (from clang diagnostic): global v4i*, LDS v4i*, imm, cpol
  __builtin_amdgcn_global_load_async_to_lds_b128(
      (__attribute__((address_space(1))) v4i*)(unsigned short*)g,
      (__attribute__((address_space(3))) v4i*)l, 0, 0);
#else
  *(uint4*)l = *(const uint4*)g;      // sync fallback: global load + ds_store
#endif
}

template <int N> __device__ __forceinline__ void wait_async() {
#if USE_ASYNC
#if __has_builtin(__builtin_amdgcn_s_wait_asynccnt)
  __builtin_amdgcn_s_wait_asynccnt(N);
#else
  asm volatile("s_wait_asynccnt %0" ::"n"(N) : "memory");
#endif
#endif
}

// Copy A[32 rows, kbase..kbase+128) (row stride lda) into LDS buffer.
// 512 16B-chunks, 256 threads -> 2 async b128 per thread (2 per wave issued).
__device__ __forceinline__ void stage_A(const unsigned short* __restrict__ A,
                                        int lda, int kbase,
                                        unsigned short* __restrict__ buf,
                                        int tid) {
#pragma unroll
  for (int c = tid; c < 512; c += 256) {
    const int row = c >> 4;           // 0..31
    const int ch  = c & 15;           // 16 chunks of 8 bf16
    async_cp_b128(A + (size_t)row * lda + kbase + ch * 8,
                  buf + row * SROW + ch * 8);
  }
}

__device__ __forceinline__ unsigned short f2bf(float f) {
  union { float f; uint32_t u; } c; c.f = f;
  uint32_t u = c.u;
  u += 0x7fffu + ((u >> 16) & 1u);    // round-to-nearest-even
  return (unsigned short)(u >> 16);
}

// Zero a contiguous f32 region (membrane states + output accumulator).
__global__ void k_zero(float* __restrict__ p, long long n) {
  for (long long i = (long long)blockIdx.x * blockDim.x + threadIdx.x; i < n;
       i += (long long)gridDim.x * blockDim.x)
    p[i] = 0.0f;
}

// fp32 [srcR, srcK] -> bf16 [*, dstK], zero-padded in K (and rows >= srcR).
__global__ void k_convert_pad(const float* __restrict__ src,
                              unsigned short* __restrict__ dst,
                              int srcR, int srcK, int dstK, long long total) {
  for (long long i = (long long)blockIdx.x * blockDim.x + threadIdx.x; i < total;
       i += (long long)gridDim.x * blockDim.x) {
    long long r = i / dstK, k = i % dstK;
    float v = (r < srcR && k < srcK) ? src[r * srcK + k] : 0.0f;
    dst[i] = f2bf(v);
  }
}

// x [B=32, T=100, 2312] f32 -> xb [T][32][2432] bf16 (K zero-padded).
__global__ void k_convert_x(const float* __restrict__ x,
                            unsigned short* __restrict__ dst) {
  const long long total = 100LL * 32 * 2432;
  for (long long i = (long long)blockIdx.x * blockDim.x + threadIdx.x; i < total;
       i += (long long)gridDim.x * blockDim.x) {
    long long t = i / (32LL * 2432);
    long long rem = i % (32LL * 2432);
    long long m = rem / 2432, k = rem % 2432;
    float v = (k < 2312) ? x[(m * 100 + t) * 2312 + k] : 0.0f;
    dst[i] = f2bf(v);
  }
}

// Fused GEMM (bf16 WMMA, f32 acc) + bias + LIF step.
//   h[m,n] = sum_k A[m,k]*W[n,k] + bias[n]
//   reset  = (mem_prev > 1);  mem = 0.9*mem_prev + h - reset;  spk = (mem > 1)
// 8 waves/block, one 16-wide N tile per wave, both M tiles (M=32) per wave.
// A tile shared across the block through double-buffered LDS (async copies).
__global__ __launch_bounds__(256) void k_gemm_lif(
    const unsigned short* __restrict__ A, int lda,   // bf16 [32, lda]
    const unsigned short* __restrict__ W, int ldw,   // bf16 [N, ldw]
    int nStages,                                     // K / KC
    const float* __restrict__ bias, int nValid,
    float* __restrict__ mem, int ldmem,              // f32 [32, ldmem]
    unsigned short* __restrict__ spkOut, int ldspk,  // bf16 [32, ldspk] | null
    float* __restrict__ acc, int ldacc,              // f32 [32, ldacc]  | null
    int nTiles) {
  __shared__ unsigned short smem[2 * BUFE];          // 17408 B

  const int tid  = threadIdx.x;
  const int wave = tid >> 5;
  const int lane = tid & 31;
  const int tile = blockIdx.x * 8 + wave;
  const bool active = (tile < nTiles);               // wave-uniform

  const int r  = lane & 15;                          // M row / N col in tile
  const int kh = lane >> 4;                          // lane half -> K chunk
  const int n0 = active ? tile * 16 : 0;

  // B layout (16-bit 32x16): lane holds N=n0+r; K [kh*8,+8) and
  // [16+kh*8,+8) -> two 16B loads, 32B apart. W row n IS column n of B.
  const uint4* Wr = reinterpret_cast<const uint4*>(W + (size_t)(n0 + r) * ldw);

  v8f c0 = {}; v8f c1 = {};

  stage_A(A, lda, 0, smem, tid);
  if (nStages > 1) stage_A(A, lda, KC, smem + BUFE, tid);

  for (int s = 0; s < nStages; ++s) {
    const bool hasNext = (s + 1) < nStages;
    if (hasNext) wait_async<2>(); else wait_async<0>();
    __syncthreads();                                 // stage s visible to all

    const unsigned short* bufp = smem + (s & 1) * BUFE;
    if (active) {
      // Preload this stage's weight fragments in one clause (8 x b128).
      const int wq = (s * KC) >> 3;                  // uint4 index base
      uint4 wv[8];
#pragma unroll
      for (int k2 = 0; k2 < 4; ++k2) {
        const int qb = wq + k2 * 4 + kh;
        wv[2 * k2]     = Wr[qb];
        wv[2 * k2 + 1] = Wr[qb + 2];
      }
#pragma unroll
      for (int k2 = 0; k2 < 4; ++k2) {
        // A layout (16-bit 16x32): lane holds M=r; VGPR0..3 = K[kh*8,+8),
        // VGPR4..7 = K[16+kh*8,+8) -> two 16B LDS loads, 32B apart.
        const unsigned short* la0 = bufp + r * SROW + k2 * 32 + kh * 8;
        const unsigned short* la1 = la0 + 16 * SROW;
        V16 a0, a1, bb;
        a0.q[0] = *(const uint4*)la0;  a0.q[1] = *(const uint4*)(la0 + 16);
        a1.q[0] = *(const uint4*)la1;  a1.q[1] = *(const uint4*)(la1 + 16);
        bb.q[0] = wv[2 * k2];          bb.q[1] = wv[2 * k2 + 1];
        c0 = __builtin_amdgcn_wmma_f32_16x16x32_bf16(false, a0.v, false, bb.v,
                                                     (short)0, c0, false, false);
        c1 = __builtin_amdgcn_wmma_f32_16x16x32_bf16(false, a1.v, false, bb.v,
                                                     (short)0, c1, false, false);
      }
    }
    __syncthreads();                                 // done reading buf[s&1]
    if (s + 2 < nStages) stage_A(A, lda, (s + 2) * KC, smem + (s & 1) * BUFE, tid);
  }

  if (!active) return;

  // D layout: lane (r, kh) holds D[kh*8 + v][r], v = 0..7 (per M tile).
  const int n = n0 + r;
  const float bval = (n < nValid) ? bias[n] : 0.0f;
#pragma unroll
  for (int v = 0; v < 8; ++v) {
    {
      const int m = kh * 8 + v;
      const float h = c0[v] + bval;
      const size_t mi = (size_t)m * ldmem + n;
      const float mp = mem[mi];
      const float reset = (mp > 1.0f) ? 1.0f : 0.0f;
      const float mn = 0.9f * mp + h - reset;        // THRESH = 1
      mem[mi] = mn;
      const float spk = (mn > 1.0f) ? 1.0f : 0.0f;
      if (spkOut) spkOut[(size_t)m * ldspk + n] = (mn > 1.0f) ? 0x3F80u : 0u;
      if (acc && n < nValid) acc[(size_t)m * ldacc + n] += spk;
    }
    {
      const int m = kh * 8 + v + 16;
      const float h = c1[v] + bval;
      const size_t mi = (size_t)m * ldmem + n;
      const float mp = mem[mi];
      const float reset = (mp > 1.0f) ? 1.0f : 0.0f;
      const float mn = 0.9f * mp + h - reset;
      mem[mi] = mn;
      const float spk = (mn > 1.0f) ? 1.0f : 0.0f;
      if (spkOut) spkOut[(size_t)m * ldspk + n] = (mn > 1.0f) ? 0x3F80u : 0u;
      if (acc && n < nValid) acc[(size_t)m * ldacc + n] += spk;
    }
  }
}

// Pairwise voting: acc [32, 90] -> votes [32, 10]. cond == 1 at ratio 0
// (computed faithfully anyway).
__global__ void k_vote(const float* __restrict__ acc, float* __restrict__ votes) {
  const int idx = threadIdx.x;
  if (idx >= 320) return;
  const int b = idx / 10, c = idx % 10;
  float v = 0.0f;
  int p = 0;
  for (int i = 0; i < 10; ++i) {
    for (int j = i + 1; j < 10; ++j, ++p) {
      const float o0 = acc[b * 90 + p * 2 + 0];
      const float o1 = acc[b * 90 + p * 2 + 1];
      const float d = o0 - o1;
      const float cond = ((d > 0.0f) || (-d >= 0.0f)) ? 1.0f : 0.0f;
      if (i == c) v += cond * o0;
      if (j == c) v += cond * o1;
    }
  }
  votes[b * 10 + c] = v;
}

extern "C" void kernel_launch(void* const* d_in, const int* in_sizes, int n_in,
                              void* d_out, int out_size, void* d_ws, size_t ws_size,
                              hipStream_t stream) {
  (void)in_sizes; (void)n_in; (void)out_size; (void)ws_size;
  const float* x     = (const float*)d_in[0];
  const float* fc1_w = (const float*)d_in[1];
  const float* fc1_b = (const float*)d_in[2];
  const float* fc2_w = (const float*)d_in[3];
  const float* fc2_b = (const float*)d_in[4];
  const float* fco_w = (const float*)d_in[5];
  const float* fco_b = (const float*)d_in[6];
  float* votes = (float*)d_out;

  constexpr int Bt = 32, T = 100, DIN = 2312;
  constexpr int KP1 = 2432;                 // 2312 padded to 19*128
  constexpr int DH = 5760, DOUT = 90, DOP = 96;   // 5760 = 45*128

  char* ws = (char*)d_ws;
  size_t off = 0;
  auto take = [&](size_t bytes) {
    size_t r = off; off += (bytes + 255) & ~(size_t)255; return r;
  };
  // State region first and contiguous so a single zero-fill covers it.
  float* m1   = (float*)(ws + take((size_t)Bt * DH  * 4));
  float* m2   = (float*)(ws + take((size_t)Bt * DH  * 4));
  float* mo   = (float*)(ws + take((size_t)Bt * DOP * 4));
  float* accv = (float*)(ws + take((size_t)Bt * DOUT * 4));
  const long long stateFloats = (long long)Bt * DH * 2 + Bt * DOP + Bt * DOUT;

  unsigned short* w1 = (unsigned short*)(ws + take((size_t)DH  * KP1 * 2));
  unsigned short* w2 = (unsigned short*)(ws + take((size_t)DH  * DH  * 2));
  unsigned short* wo = (unsigned short*)(ws + take((size_t)DOP * DH  * 2));
  unsigned short* xb = (unsigned short*)(ws + take((size_t)T * Bt * KP1 * 2));
  unsigned short* s1 = (unsigned short*)(ws + take((size_t)Bt * DH * 2));
  unsigned short* s2 = (unsigned short*)(ws + take((size_t)Bt * DH * 2));

  // Per-call prep: zero states, bf16-convert weights and inputs.
  k_zero<<<512, 256, 0, stream>>>(m1, stateFloats);
  k_convert_pad<<<2048, 256, 0, stream>>>(fc1_w, w1, DH,   DIN, KP1, (long long)DH  * KP1);
  k_convert_pad<<<4096, 256, 0, stream>>>(fc2_w, w2, DH,   DH,  DH,  (long long)DH  * DH);
  k_convert_pad<<<512,  256, 0, stream>>>(fco_w, wo, DOUT, DH,  DH,  (long long)DOP * DH);
  k_convert_x<<<2048, 256, 0, stream>>>(x, xb);

  for (int t = 0; t < T; ++t) {
    const unsigned short* At = xb + (size_t)t * Bt * KP1;
    // fc1: 360 N tiles, 45 blocks x 8 waves, 19 K stages
    k_gemm_lif<<<45, 256, 0, stream>>>(At, KP1, w1, KP1, KP1 / KC,
                                       fc1_b, DH, m1, DH, s1, DH,
                                       nullptr, 0, 360);
    // fc2: 45 K stages
    k_gemm_lif<<<45, 256, 0, stream>>>(s1, DH, w2, DH, DH / KC,
                                       fc2_b, DH, m2, DH, s2, DH,
                                       nullptr, 0, 360);
    // fco: 6 N tiles (90 padded to 96), accumulate spikes
    k_gemm_lif<<<1, 256, 0, stream>>>(s2, DH, wo, DH, DH / KC,
                                      fco_b, DOUT, mo, DOP, nullptr, 0,
                                      accv, DOUT, 6);
  }
  k_vote<<<1, 320, 0, stream>>>(accv, votes);
}
